// multi_head_self_attention_17454747090977
// MI455X (gfx1250) — matrix operand order repaired and checked by reading, not compiled, not run on hardware
//
#include <hip/hip_runtime.h>

// MI455X / gfx1250: wave32, WMMA bf16 16x16x32 with f32 accumulation.

typedef __attribute__((ext_vector_type(16))) __bf16 v16bf;
typedef __attribute__((ext_vector_type(8)))  __bf16 v8bf;
typedef __attribute__((ext_vector_type(8)))  float  v8f;

#define B_  4
#define S_  2048
#define E_  1024
#define D_  64
#define H_  16
#define KC_ 4          // split-K chunks in the AV GEMM

// ---------------------------------------------------------------------------
// Kernel 0: repack Wq/Wk/Wv (f32 [E,64]) into bf16 WMMA B-fragment layout.
// pack[w][ks][nt][lane][e]; B-frag striping (ISA 7.12.2): lanes 0-15 hold
// K {0-7,16-23}, lanes 16-31 K {8-15,24-31} of a 32-deep step; lane&15 = column in tile.
// ---------------------------------------------------------------------------
__global__ __launch_bounds__(256) void pack_weights(
    const float* __restrict__ Wq, const float* __restrict__ Wk,
    const float* __restrict__ Wv, __bf16* __restrict__ wpack) {
  int t = blockIdx.x * blockDim.x + threadIdx.x;   // 0 .. 3*65536-1
  int w  = t >> 16;
  int r  = t & 65535;                              // ((ks*4+nt)*32+lane)*16+e
  int ks = r >> 11;
  int r2 = r & 2047;
  int nt   = r2 >> 9;
  int lane = (r2 >> 4) & 31;
  int e    = r2 & 15;
  int col = nt * 16 + (lane & 15);
  int k   = ks * 32 + ((lane & 16) ? 8 : 0) + (e < 8 ? e : e + 8);
  const float* W = (w == 0) ? Wq : (w == 1) ? Wk : Wv;
  wpack[t] = (__bf16)W[k * D_ + col];
}

// ---------------------------------------------------------------------------
// Kernel 1: Q/K/V projection. One wave -> one 16x64 output tile of one matrix.
// A-frag (16x32 bf16): lane&15 = row; lanes<16 get K {0-7,16-23}, lanes>=16
// get K {8-15,24-31} -> two contiguous 8-float chunks, cvt to bf16.
// Q,K stored row-major bf16 [S,64]; V stored transposed bf16 [64,S].
// ---------------------------------------------------------------------------
__global__ __launch_bounds__(256) void qkv_project(
    const float* __restrict__ x, const __bf16* __restrict__ wpack,
    __bf16* __restrict__ Qb, __bf16* __restrict__ Kb, __bf16* __restrict__ Vt) {
  const int lane  = threadIdx.x & 31;
  const int gwave = (blockIdx.x * blockDim.x + threadIdx.x) >> 5;  // 0..1535
  const int which = gwave >> 9;            // 0=Q 1=K 2=V
  const int rem   = gwave & 511;
  const int b     = rem >> 7;
  const int mt    = rem & 127;

  const int half8 = (lane & 16) ? 8 : 0;
  const int row   = mt * 16 + (lane & 15);
  const float* xrow = x + (size_t)(b * S_ + row) * E_;
  const __bf16* wp  = wpack + which * 65536;

  v8f acc[4];
  #pragma unroll
  for (int nt = 0; nt < 4; ++nt)
    #pragma unroll
    for (int e = 0; e < 8; ++e) acc[nt][e] = 0.f;

  for (int ks = 0; ks < 32; ++ks) {
    v16bf a;
    const float* p0 = xrow + ks * 32 + half8;
    const float* p1 = p0 + 16;
    #pragma unroll
    for (int e = 0; e < 8; ++e) { a[e] = (__bf16)p0[e]; a[8 + e] = (__bf16)p1[e]; }
    #pragma unroll
    for (int nt = 0; nt < 4; ++nt) {
      v16bf bf = *(const v16bf*)(wp + (((ks * 4 + nt) * 32 + lane) << 4));
      acc[nt] = __builtin_amdgcn_wmma_f32_16x16x32_bf16(
          false, a, false, bf, (short)0, acc[nt], false, false);
    }
  }

  // C/D layout: VGPR r -> row (lane<16 ? r : 8+r), col = lane&15 within tile.
  const int rowbase = mt * 16 + ((lane & 16) ? 8 : 0);
  const int col     = lane & 15;
  if (which == 2) {
    #pragma unroll
    for (int nt = 0; nt < 4; ++nt)
      #pragma unroll
      for (int r = 0; r < 8; ++r)
        Vt[(size_t)(b * D_ + nt * 16 + col) * S_ + (rowbase + r)] = (__bf16)acc[nt][r];
  } else {
    __bf16* dst = (which == 0) ? Qb : Kb;
    #pragma unroll
    for (int nt = 0; nt < 4; ++nt)
      #pragma unroll
      for (int r = 0; r < 8; ++r)
        dst[(size_t)(b * S_ + rowbase + r) * D_ + nt * 16 + col] = (__bf16)acc[nt][r];
  }
}

// ---------------------------------------------------------------------------
// Kernel 2: att = Q @ K^T (contract over d). B-frag of K^T = contiguous rows
// of K. One wave -> 16 x 128 tile of att (8 accumulators), K-loop = 2.
// ---------------------------------------------------------------------------
__global__ __launch_bounds__(256) void scores(
    const __bf16* __restrict__ Qb, const __bf16* __restrict__ Kb,
    float* __restrict__ att) {
  const int lane  = threadIdx.x & 31;
  const int gwave = (blockIdx.x * blockDim.x + threadIdx.x) >> 5;  // 0..8191
  const int b   = gwave >> 11;
  const int rem = gwave & 2047;
  const int mt  = rem >> 4;    // 128 row tiles
  const int ng  = rem & 15;    // 16 groups of 8 col tiles

  const int row    = mt * 16 + (lane & 15);
  const int half8  = (lane & 16) ? 8 : 0;
  const int half16 = (lane & 16) ? 16 : 0;
  const __bf16* qrow = Qb + (size_t)(b * S_ + row) * D_;

  v8f acc[8];
  #pragma unroll
  for (int t = 0; t < 8; ++t)
    #pragma unroll
    for (int e = 0; e < 8; ++e) acc[t][e] = 0.f;

  #pragma unroll
  for (int ks = 0; ks < 2; ++ks) {
    v16bf a;
    v8bf lo = *(const v8bf*)(qrow + ks * 32 + half16);
    v8bf hi = *(const v8bf*)(qrow + ks * 32 + half16 + 8);
    #pragma unroll
    for (int e = 0; e < 8; ++e) { a[e] = lo[e]; a[8 + e] = hi[e]; }
    #pragma unroll
    for (int t = 0; t < 8; ++t) {
      const int j = (ng * 8 + t) * 16 + (lane & 15);
      v16bf bf = *(const v16bf*)(Kb + (size_t)(b * S_ + j) * D_ + ks * 32 + half16);
      acc[t] = __builtin_amdgcn_wmma_f32_16x16x32_bf16(
          false, a, false, bf, (short)0, acc[t], false, false);
    }
  }

  const int rowbase = mt * 16 + ((lane & 16) ? 8 : 0);
  #pragma unroll
  for (int t = 0; t < 8; ++t) {
    const int j = (ng * 8 + t) * 16 + (lane & 15);
    #pragma unroll
    for (int r = 0; r < 8; ++r)
      att[(size_t)(b * S_ + rowbase + r) * S_ + j] = acc[t][r];
  }
}

// ---------------------------------------------------------------------------
// Kernel 3: softmax over the QUERY axis (reference quirk: Softmax(dim=1)).
// One thread per (batch, key column); stride-S accesses coalesce across
// adjacent threads. ONLINE max+sum in pass 1 (flash-style rescale) folds the
// usual 3 memory passes into 2 -> saves one full 67 MB sweep of att.
// ---------------------------------------------------------------------------
__global__ __launch_bounds__(256) void softmax_axis_q(float* __restrict__ att) {
  const int t = blockIdx.x * blockDim.x + threadIdx.x;  // 0..8191
  const int b = t >> 11;
  const int k = t & 2047;
  float* p = att + (size_t)b * S_ * S_ + k;
  float m = -3.402823466e38f;
  float sum = 0.f;
  for (int i = 0; i < S_; ++i) {
    const float v  = p[(size_t)i * S_];
    const float nm = fmaxf(m, v);
    sum = sum * __expf(m - nm) + __expf(v - nm);
    m = nm;
  }
  const float inv = 1.f / sum;
  for (int i = 0; i < S_; ++i)
    p[(size_t)i * S_] = __expf(p[(size_t)i * S_] - m) * inv;
}

// ---------------------------------------------------------------------------
// Kernel 4: split-K partial of out = att @ V. Wave -> (b, mt, kc): 16x64 tile
// accumulated over 512 of the 2048 K values (16 WMMA steps). 2048 waves keep
// the 67 MB att stream at full HBM concurrency. Partials are written to
// per-chunk buffers (deterministic: no float atomics).
// ---------------------------------------------------------------------------
__global__ __launch_bounds__(256) void av_partial(
    const float* __restrict__ att, const __bf16* __restrict__ Vt,
    float* __restrict__ partial) {
  const int lane  = threadIdx.x & 31;
  const int gwave = (blockIdx.x * blockDim.x + threadIdx.x) >> 5;  // 0..2047
  const int kc  = gwave >> 9;          // 0..3
  const int rem = gwave & 511;
  const int b   = rem >> 7;
  const int mt  = rem & 127;

  const int row    = mt * 16 + (lane & 15);
  const int half8  = (lane & 16) ? 8 : 0;
  const int half16 = (lane & 16) ? 16 : 0;
  const float* arow = att + (size_t)(b * S_ + row) * S_;

  v8f acc[4];
  #pragma unroll
  for (int nt = 0; nt < 4; ++nt)
    #pragma unroll
    for (int e = 0; e < 8; ++e) acc[nt][e] = 0.f;

  const int ks0 = kc * 16;
  for (int ks = ks0; ks < ks0 + 16; ++ks) {
    v16bf a;
    const float* p0 = arow + ks * 32 + half16;
    const float* p1 = p0 + 8;
    #pragma unroll
    for (int e = 0; e < 8; ++e) { a[e] = (__bf16)p0[e]; a[8 + e] = (__bf16)p1[e]; }
    #pragma unroll
    for (int nt = 0; nt < 4; ++nt) {
      const int d = nt * 16 + (lane & 15);
      v16bf bf = *(const v16bf*)(Vt + (size_t)(b * D_ + d) * S_ + ks * 32 + half16);
      acc[nt] = __builtin_amdgcn_wmma_f32_16x16x32_bf16(
          false, a, false, bf, (short)0, acc[nt], false, false);
    }
  }

  float* pk = partial + (size_t)kc * (B_ * S_ * D_);
  const int rowbase = mt * 16 + ((lane & 16) ? 8 : 0);
  #pragma unroll
  for (int nt = 0; nt < 4; ++nt) {
    const int d = nt * 16 + (lane & 15);
    #pragma unroll
    for (int r = 0; r < 8; ++r)
      pk[(size_t)(b * S_ + rowbase + r) * D_ + d] = acc[nt][r];
  }
}

// ---------------------------------------------------------------------------
// Kernel 5: combine the 4 split-K partials and fan out the x16 head tiling.
// One thread per (b,s,d); reads 4 floats, writes 16 copies (coalesced in d).
// ---------------------------------------------------------------------------
__global__ __launch_bounds__(256) void combine_out(
    const float* __restrict__ partial, float* __restrict__ out) {
  const int t = blockIdx.x * blockDim.x + threadIdx.x;  // 0 .. B*S*D-1
  const int b   = t >> 17;
  const int rem = t & 131071;
  const int s   = rem >> 6;
  const int d   = rem & 63;
  const size_t idx = (size_t)(b * S_ + s) * D_ + d;
  float v = 0.f;
  #pragma unroll
  for (int kc = 0; kc < KC_; ++kc)
    v += partial[(size_t)kc * (B_ * S_ * D_) + idx];
  float* orow = out + (size_t)(b * S_ + s) * (D_ * H_);
  #pragma unroll
  for (int h = 0; h < H_; ++h) orow[h * 64 + d] = v;
}

// ---------------------------------------------------------------------------
extern "C" void kernel_launch(void* const* d_in, const int* in_sizes, int n_in,
                              void* d_out, int out_size, void* d_ws, size_t ws_size,
                              hipStream_t stream) {
  const float* x  = (const float*)d_in[0];
  const float* Wq = (const float*)d_in[1];
  const float* Wk = (const float*)d_in[2];
  const float* Wv = (const float*)d_in[3];
  float* out = (float*)d_out;

  char* ws = (char*)d_ws;
  float*  att = (float*)ws;                       // 4*2048*2048*4 = 67,108,864 B
  float*  par = (float*)(ws + 67108864);          // 4 * 2 MB split-K partials
  __bf16* Qb  = (__bf16*)(ws + 75497472);         // 1 MB
  __bf16* Kb  = (__bf16*)(ws + 76546048);         // 1 MB
  __bf16* Vt  = (__bf16*)(ws + 77594624);         // 1 MB (transposed [D,S])
  __bf16* wpk = (__bf16*)(ws + 78643200);         // 3 * 128 KB packed weights

  pack_weights  <<<768,  256, 0, stream>>>(Wq, Wk, Wv, wpk);
  qkv_project   <<<192,  256, 0, stream>>>(x, wpk, Qb, Kb, Vt);
  scores        <<<1024, 256, 0, stream>>>(Qb, Kb, att);
  softmax_axis_q<<<32,   256, 0, stream>>>(att);
  av_partial    <<<256,  256, 0, stream>>>(att, Vt, par);
  combine_out   <<<2048, 256, 0, stream>>>(par, out);
}
